// SequenceHead_67405216743687
// MI455X (gfx1250) — matrix operand order repaired and checked by reading, not compiled, not run on hardware
//
#include <hip/hip_runtime.h>
#include <hip/hip_bf16.h>

typedef __attribute__((ext_vector_type(16))) _Float16 v16h;
typedef __attribute__((ext_vector_type(8)))  _Float16 h8;
typedef __attribute__((ext_vector_type(8)))  float    v8f;

#define N_NODES  4000
#define N_EDGES  64000
#define C0       32
#define C1       16
#define H_DIM    64
#define NUM_AA   20
#define EDGE_DIM 17
#define MID      32

#define EPB      32     // edges per block (2 WMMA M-tiles)

// -------- workspace layout (byte offsets, base assumed >=256B aligned) -----
#define WS_NODEACC 0                                    // 4000*64 f32 = 1,024,000 B
#define WS_W3A     (N_NODES * H_DIM * 4)                // 32*2048 f16 = 131,072 B
#define WS_W3B     (WS_W3A + MID * H_DIM * C0 * 2)      // 32*1024 f16 =  65,536 B
#define WS_KS      (WS_W3B + MID * H_DIM * C1 * 2)      // 64*32  f16 =    4,096 B

// force both WMMA A-operands into distinct live VGPR blocks so the
// other tile's pk_mul build fills the WMMA WAR-hazard co-exec slots
#define PIN2(a, b) asm volatile("" :: "v"(a), "v"(b))

// ---------------------------------------------------------------------------
// Kernel 0: zero node accumulator + convert weights to f16
// ---------------------------------------------------------------------------
__global__ void prep_kernel(const float* __restrict__ W3a,
                            const float* __restrict__ W3b,
                            const float* __restrict__ Ks,
                            float* __restrict__ node_acc,
                            _Float16* __restrict__ W3a_h,
                            _Float16* __restrict__ W3b_h,
                            _Float16* __restrict__ Ks_h) {
    int i = blockIdx.x * 256 + threadIdx.x;
    if (i < N_NODES * H_DIM) node_acc[i] = 0.0f;
    if (i < MID * H_DIM * C0) W3a_h[i] = (_Float16)W3a[i];
    if (i < MID * H_DIM * C1) W3b_h[i] = (_Float16)W3b[i];
    if (i < H_DIM * C0)       Ks_h[i]  = (_Float16)Ks[i];
}

// ---------------------------------------------------------------------------
// Kernel 1: per-edge fused radial-MLP + WMMA contraction + atomic segment_sum
// 1 WG = 128 threads (4 wave32) handles 32 edges; wave w owns heads [16w,16w+16)
// and computes TWO 16x16 fp32 tiles sharing every B (weight) fetch.
// ---------------------------------------------------------------------------
__global__ __launch_bounds__(128)
void edge_kernel(const float* __restrict__ feats0,      // (4000,32)
                 const float* __restrict__ feats1,      // (4000,16,3)
                 const float* __restrict__ edge_feats,  // (64000,17)
                 const int*   __restrict__ src,
                 const int*   __restrict__ dst,
                 const float* __restrict__ basis,       // (64000,3)
                 const float* __restrict__ W1a, const float* __restrict__ b1a,
                 const float* __restrict__ W2a, const float* __restrict__ b2a,
                 const float* __restrict__ W1b, const float* __restrict__ b1b,
                 const float* __restrict__ W2b, const float* __restrict__ b2b,
                 const _Float16* __restrict__ W3a_h,    // (32, 64*32)
                 const _Float16* __restrict__ W3b_h,    // (32, 64*16)
                 const _Float16* __restrict__ Ks_h,     // (64, 32)
                 float* __restrict__ node_acc) {        // (4000, 64)
    __shared__ float s_inv[EPB * 20];
    __shared__ float s_h1 [EPB * MID];
    __shared__ __align__(16) _Float16 s_h2a[EPB * MID];
    __shared__ __align__(16) _Float16 s_h2b[EPB * MID];
    __shared__ __align__(16) _Float16 s_f0 [EPB * C0];
    __shared__ __align__(16) _Float16 s_fd [EPB * C0];
    __shared__ __align__(16) _Float16 s_tmp[EPB * C1];
    __shared__ int s_src[EPB], s_dst[EPB];

    const int tid = threadIdx.x;
    const int E0  = blockIdx.x * EPB;

    // ---- stage edge indices + invariant edge features -----------------
    if (tid < EPB) { s_src[tid] = src[E0 + tid]; s_dst[tid] = dst[E0 + tid]; }
    for (int idx = tid; idx < EPB * EDGE_DIM; idx += 128) {
        int e = idx / EDGE_DIM, i = idx % EDGE_DIM;
        s_inv[e * 20 + i] = edge_feats[(E0 + e) * EDGE_DIM + i];
    }
    __syncthreads();

    // ---- gather node features (f16) + basis contraction ---------------
    for (int idx = tid; idx < EPB * C0; idx += 128) {
        int e = idx >> 5, c = idx & 31;
        s_f0[e * C0 + c] = (_Float16)feats0[s_src[e] * C0 + c];
        s_fd[e * C0 + c] = (_Float16)feats0[s_dst[e] * C0 + c];
    }
    for (int idx = tid; idx < EPB * C1; idx += 128) {
        int e = idx >> 4, c = idx & 15;
        const float* f1 = feats1 + (s_src[e] * C1 + c) * 3;
        const float* bs = basis + (E0 + e) * 3;
        s_tmp[e * C1 + c] = (_Float16)(f1[0]*bs[0] + f1[1]*bs[1] + f1[2]*bs[2]);
    }
    __syncthreads();

    // ---- radial MLP "a": 17 -> 32 -> 32 (relu) -------------------------
    for (int idx = tid; idx < EPB * MID; idx += 128) {
        int e = idx >> 5, j = idx & 31;
        float a = b1a[j];
        #pragma unroll
        for (int i = 0; i < EDGE_DIM; ++i) a += s_inv[e*20+i] * W1a[i*MID + j];
        s_h1[e * MID + j] = fmaxf(a, 0.0f);
    }
    __syncthreads();
    for (int idx = tid; idx < EPB * MID; idx += 128) {
        int e = idx >> 5, j = idx & 31;
        float a = b2a[j];
        #pragma unroll
        for (int i = 0; i < MID; ++i) a += s_h1[e*MID+i] * W2a[i*MID + j];
        s_h2a[e * MID + j] = (_Float16)fmaxf(a, 0.0f);
    }
    __syncthreads();
    // ---- radial MLP "b" (reuses s_h1) ----------------------------------
    for (int idx = tid; idx < EPB * MID; idx += 128) {
        int e = idx >> 5, j = idx & 31;
        float a = b1b[j];
        #pragma unroll
        for (int i = 0; i < EDGE_DIM; ++i) a += s_inv[e*20+i] * W1b[i*MID + j];
        s_h1[e * MID + j] = fmaxf(a, 0.0f);
    }
    __syncthreads();
    for (int idx = tid; idx < EPB * MID; idx += 128) {
        int e = idx >> 5, j = idx & 31;
        float a = b2b[j];
        #pragma unroll
        for (int i = 0; i < MID; ++i) a += s_h1[e*MID+i] * W2b[i*MID + j];
        s_h2b[e * MID + j] = (_Float16)fmaxf(a, 0.0f);
    }
    __syncthreads();

    // ---- WMMA phase: each wave -> 2 tiles of 16 edges x 16 heads -------
    const int wv    = tid >> 5;
    const int lane  = tid & 31;
    const int n     = lane & 15;      // A-row (M) / B-col (N) owned by this lane
    const int selHi = lane >> 4;      // lane-half selects K sub-range (ISA 7.12.2)
    const int hbase = wv * 16;
    const int e0 = n, e1 = n + 16;    // A-matrix rows for tile0 / tile1

    // A-operand f16 layout, 16x32: lanes<16 hold K={0..7,16..23}, lanes>=16 K={8..15,24..31}
    const h8 f0lo0 = *(const h8*)(s_f0 + e0 * C0 + selHi * 8);
    const h8 f0hi0 = *(const h8*)(s_f0 + e0 * C0 + 16 + selHi * 8);
    const h8 f0lo1 = *(const h8*)(s_f0 + e1 * C0 + selHi * 8);
    const h8 f0hi1 = *(const h8*)(s_f0 + e1 * C0 + 16 + selHi * 8);
    const h8 tmp0  = *(const h8*)(s_tmp + e0 * C1 + selHi * 8);
    const h8 tmp1  = *(const h8*)(s_tmp + e1 * C1 + selHi * 8);

    v8f acc0 = {}, acc1 = {};

    // ===== part a: K = 32 m-steps, A = outer(h2a[:,m], f0) ==============
    const _Float16* h2aR0 = s_h2a + e0 * MID;
    const _Float16* h2aR1 = s_h2a + e1 * MID;
    for (int m = 0; m < MID; ++m) {
        _Float16 sa0 = h2aR0[m], sa1 = h2aR1[m];
        v16h A0, A1;
        #pragma unroll
        for (int i = 0; i < 8; ++i) {
            A0[i] = f0lo0[i] * sa0; A0[i+8] = f0hi0[i] * sa0;
            A1[i] = f0lo1[i] * sa1; A1[i+8] = f0hi1[i] * sa1;
        }
        PIN2(A0, A1);
        // B layout 32x16: lanes<16 hold K={0..7,16..23}, lanes>=16 hold K={8..15,24..31} (col n)
        const _Float16* bb = W3a_h + m * (H_DIM*C0) + (hbase + n) * C0 + selHi * 8;
        __builtin_prefetch((const void*)(bb + H_DIM*C0), 0, 1);   // next m row
        h8 b0 = *(const h8*)bb;
        h8 b1 = *(const h8*)(bb + 16);
        v16h B = __builtin_shufflevector(b0, b1, 0,1,2,3,4,5,6,7,8,9,10,11,12,13,14,15);
        acc0 = __builtin_amdgcn_wmma_f32_16x16x32_f16(false, A0, false, B,
                                                      (short)0, acc0, false, false);
        acc1 = __builtin_amdgcn_wmma_f32_16x16x32_f16(false, A1, false, B,
                                                      (short)0, acc1, false, false);
    }

    // ===== part b: K = 512 (2 m-rows per WMMA, C1=16) ===================
    const _Float16* h2bR0 = s_h2b + e0 * MID;
    const _Float16* h2bR1 = s_h2b + e1 * MID;
    for (int mb = 0; mb < MID; mb += 2) {
        _Float16 s00 = h2bR0[mb], s01 = h2bR0[mb + 1];
        _Float16 s10 = h2bR1[mb], s11 = h2bR1[mb + 1];
        v16h A0, A1;
        #pragma unroll
        for (int i = 0; i < 8; ++i) {
            A0[i] = tmp0[i] * s00; A0[i+8] = tmp0[i] * s01;
            A1[i] = tmp1[i] * s10; A1[i+8] = tmp1[i] * s11;
        }
        PIN2(A0, A1);
        const _Float16* bb = W3b_h + mb * (H_DIM*C1) + (hbase + n) * C1 + selHi * 8;
        __builtin_prefetch((const void*)(bb + 2 * H_DIM*C1), 0, 1);
        h8 b0 = *(const h8*)bb;
        h8 b1 = *(const h8*)(bb + H_DIM*C1);
        v16h B = __builtin_shufflevector(b0, b1, 0,1,2,3,4,5,6,7,8,9,10,11,12,13,14,15);
        acc0 = __builtin_amdgcn_wmma_f32_16x16x32_f16(false, A0, false, B,
                                                      (short)0, acc0, false, false);
        acc1 = __builtin_amdgcn_wmma_f32_16x16x32_f16(false, A1, false, B,
                                                      (short)0, acc1, false, false);
    }

    // ===== self term: out += f0[dst] @ kernel_self^T (K = 32) ===========
    {
        h8 d00 = *(const h8*)(s_fd + e0 * C0 + selHi * 8);
        h8 d01 = *(const h8*)(s_fd + e0 * C0 + 16 + selHi * 8);
        h8 d10 = *(const h8*)(s_fd + e1 * C0 + selHi * 8);
        h8 d11 = *(const h8*)(s_fd + e1 * C0 + 16 + selHi * 8);
        v16h A0 = __builtin_shufflevector(d00, d01, 0,1,2,3,4,5,6,7,8,9,10,11,12,13,14,15);
        v16h A1 = __builtin_shufflevector(d10, d11, 0,1,2,3,4,5,6,7,8,9,10,11,12,13,14,15);
        PIN2(A0, A1);
        const _Float16* kb = Ks_h + (hbase + n) * C0 + selHi * 8;
        h8 b0 = *(const h8*)kb;
        h8 b1 = *(const h8*)(kb + 16);
        v16h B = __builtin_shufflevector(b0, b1, 0,1,2,3,4,5,6,7,8,9,10,11,12,13,14,15);
        acc0 = __builtin_amdgcn_wmma_f32_16x16x32_f16(false, A0, false, B,
                                                      (short)0, acc0, false, false);
        acc1 = __builtin_amdgcn_wmma_f32_16x16x32_f16(false, A1, false, B,
                                                      (short)0, acc1, false, false);
    }

    // ---- scatter-add into node accumulator (segment_sum over dst) -----
    // C layout: lanes<16: (M=r, N=lane); lanes>=16: (M=r+8, N=lane-16)
    #pragma unroll
    for (int r = 0; r < 8; ++r) {
        int Mr = r + selHi * 8;
        atomicAdd(node_acc + s_dst[Mr]      * H_DIM + hbase + n, acc0[r]);
        atomicAdd(node_acc + s_dst[Mr + 16] * H_DIM + hbase + n, acc1[r]);
    }
}

// ---------------------------------------------------------------------------
// Kernel 2: node head MLP + log_softmax. 1 WG (64 threads) per node.
// ---------------------------------------------------------------------------
__global__ __launch_bounds__(64)
void node_kernel(const float* __restrict__ node_acc,
                 const float* __restrict__ Wm1, const float* __restrict__ bm1,
                 const float* __restrict__ Wm2, const float* __restrict__ bm2,
                 float* __restrict__ out) {
    __shared__ float h[H_DIM];
    __shared__ float p1[H_DIM];
    __shared__ float p2[NUM_AA];
    __shared__ float norm;
    const int nid = blockIdx.x, t = threadIdx.x;

    h[t] = node_acc[nid * H_DIM + t];
    __syncthreads();
    float a = bm1[t];
    #pragma unroll 8
    for (int i = 0; i < H_DIM; ++i) a += h[i] * Wm1[i * H_DIM + t];
    p1[t] = fmaxf(a, 0.0f);
    __syncthreads();
    if (t < NUM_AA) {
        float b = bm2[t];
        #pragma unroll 8
        for (int i = 0; i < H_DIM; ++i) b += p1[i] * Wm2[i * NUM_AA + t];
        p2[t] = b;
    }
    __syncthreads();
    if (t == 0) {
        float mx = p2[0];
        for (int j = 1; j < NUM_AA; ++j) mx = fmaxf(mx, p2[j]);
        float s = 0.0f;
        for (int j = 0; j < NUM_AA; ++j) s += __expf(p2[j] - mx);
        norm = mx + __logf(s);
    }
    __syncthreads();
    if (t < NUM_AA) out[nid * NUM_AA + t] = p2[t] - norm;
}

// ---------------------------------------------------------------------------
extern "C" void kernel_launch(void* const* d_in, const int* in_sizes, int n_in,
                              void* d_out, int out_size, void* d_ws, size_t ws_size,
                              hipStream_t stream) {
    const float* feats0 = (const float*)d_in[0];
    const float* feats1 = (const float*)d_in[1];
    const float* edgef  = (const float*)d_in[2];
    const int*   src    = (const int*)d_in[3];
    const int*   dst    = (const int*)d_in[4];
    const float* basis  = (const float*)d_in[5];
    const float* W1a = (const float*)d_in[6];  const float* b1a = (const float*)d_in[7];
    const float* W2a = (const float*)d_in[8];  const float* b2a = (const float*)d_in[9];
    const float* W3a = (const float*)d_in[10];
    const float* W1b = (const float*)d_in[11]; const float* b1b = (const float*)d_in[12];
    const float* W2b = (const float*)d_in[13]; const float* b2b = (const float*)d_in[14];
    const float* W3b = (const float*)d_in[15];
    const float* Ks  = (const float*)d_in[16];
    const float* Wm1 = (const float*)d_in[17]; const float* bm1 = (const float*)d_in[18];
    const float* Wm2 = (const float*)d_in[19]; const float* bm2 = (const float*)d_in[20];
    float* out = (float*)d_out;

    char* ws = (char*)d_ws;
    float*    node_acc = (float*)(ws + WS_NODEACC);
    _Float16* W3a_h    = (_Float16*)(ws + WS_W3A);
    _Float16* W3b_h    = (_Float16*)(ws + WS_W3B);
    _Float16* Ks_h     = (_Float16*)(ws + WS_KS);

    prep_kernel<<<(N_NODES * H_DIM + 255) / 256, 256, 0, stream>>>(
        W3a, W3b, Ks, node_acc, W3a_h, W3b_h, Ks_h);

    edge_kernel<<<N_EDGES / EPB, 128, 0, stream>>>(
        feats0, feats1, edgef, src, dst, basis,
        W1a, b1a, W2a, b2a, W1b, b1b, W2b, b2b,
        W3a_h, W3b_h, Ks_h, node_acc);

    node_kernel<<<N_NODES, H_DIM, 0, stream>>>(node_acc, Wm1, bm1, Wm2, bm2, out);
}
